// SpatialPoolingAttention_80307298501265
// MI455X (gfx1250) — compile-verified, matrix operand order checked by reading, not run on hardware
//
#include <hip/hip_runtime.h>
#include <hip/hip_bf16.h>

#define NTOK 16
#define CDIM 512
#define HD   64
#define MPAD 32   // pools rows padded 28 -> 32

typedef __attribute__((ext_vector_type(16))) __bf16 v16bf;
typedef __attribute__((ext_vector_type(8)))  float  v8f;
typedef __attribute__((ext_vector_type(8)))  unsigned short u16x8;

union V16 { v16bf bf; u16x8 h[2]; unsigned short s[16]; };

__constant__ int G1[16] = {2, 3, 5, 6, 1, 4, 0, 7, 8, 9, 14, 15, 11, 12, 10, 13};
__constant__ int G2[8]  = {0, 1, 2, 3, 5, 6, 4, 7};

// hardware bf16 convert (RNE); backend packs adjacent pairs
__device__ __forceinline__ unsigned short f2bf(float f) {
  const __bf16 h = (__bf16)f;
  return __builtin_bit_cast(unsigned short, h);
}

__device__ __forceinline__ v8f wmma_bf16(const V16& a, const V16& b, v8f c) {
  return __builtin_amdgcn_wmma_f32_16x16x32_bf16(false, a.bf, false, b.bf,
                                                 (short)0, c, false, false);
}

// A-operand (or same-pattern B for K-major storage): 8 contig bf16 at p, 8 at p+16
__device__ __forceinline__ V16 ldsA(const unsigned short* p) {
  V16 r; r.h[0] = *(const u16x8*)p; r.h[1] = *(const u16x8*)(p + 16); return r;
}
// Packed-weight B-operand: 16 contiguous bf16 per lane
__device__ __forceinline__ V16 ldgB(const unsigned short* p) {
  V16 r; r.h[0] = *(const u16x8*)p; r.h[1] = *(const u16x8*)(p + 8); return r;
}
__device__ __forceinline__ void ld16(const float* __restrict__ p, float* v) {
  const float4 a = *(const float4*)(p);
  const float4 b = *(const float4*)(p + 4);
  const float4 c = *(const float4*)(p + 8);
  const float4 d = *(const float4*)(p + 12);
  v[0]=a.x; v[1]=a.y; v[2]=a.z;  v[3]=a.w;  v[4]=b.x;  v[5]=b.y;  v[6]=b.z;  v[7]=b.w;
  v[8]=c.x; v[9]=c.y; v[10]=c.z; v[11]=c.w; v[12]=d.x; v[13]=d.y; v[14]=d.z; v[15]=d.w;
}

// ---- pack Wq (512x512), Wkv (512x1024), Wproj (512x512) into bf16 WMMA-B layout ----
// packed idx: (((ntile*16 + kb)*32 + lane)*16 + e); lane half=l>>4, nl=l&15
// value = W[(kb*32 + kidx(e,half)) * N + ntile*16 + nl]
__global__ void pack_weights(const float* __restrict__ Wq, const float* __restrict__ Wkv,
                             const float* __restrict__ Wpr, unsigned short* __restrict__ wp) {
  const unsigned t = blockIdx.x * 256u + threadIdx.x;
  const float* W; unsigned N, idx;
  if (t < 262144u)      { W = Wq;  N = 512;  idx = t; }
  else if (t < 786432u) { W = Wkv; N = 1024; idx = t - 262144u; }
  else                  { W = Wpr; N = 512;  idx = t - 786432u; }
  const unsigned e = idx & 15u, lane = (idx >> 4) & 31u;
  const unsigned kb = (idx >> 9) & 15u, nt = idx >> 13;
  const unsigned half = lane >> 4, nl = lane & 15u;
  const unsigned kidx = (e < 8u) ? (half * 8u + e) : (16u + half * 8u + (e - 8u));
  wp[t] = f2bf(W[(kb * 32u + kidx) * N + nt * 16u + nl]);
}

__global__ __launch_bounds__(256) void attn_fused(
    const float* __restrict__ x, const unsigned short* __restrict__ wp,
    const float* __restrict__ bproj, const float* __restrict__ gamma,
    const float* __restrict__ beta, float* __restrict__ out) {
  __shared__ unsigned short sP[MPAD * CDIM];   // 32KB: LN'd pools bf16 -> per-wave k/P scratch
  __shared__ unsigned short sQO[NTOK * CDIM];  // 16KB: per-wave q bf16 -> O bf16
  __shared__ unsigned short sX[NTOK * CDIM];   // 16KB: raw x bf16 (q-GEMM A operand)

  const int lane = threadIdx.x & 31;
  const int w    = threadIdx.x >> 5;           // wave == head (8 waves, 8 heads)
  const int half = lane >> 4;
  const int nl   = lane & 15;
  const int b    = blockIdx.x;

  const unsigned short* wqP  = wp;
  const unsigned short* wkvP = wp + 262144;
  const unsigned short* wprP = wp + 786432;
  const float* xb = x + (size_t)b * NTOK * CDIM;

  const v8f vzero = {0.f, 0.f, 0.f, 0.f, 0.f, 0.f, 0.f, 0.f};

  // ---------- Stage 1: pools (16 raw + 8 G1-pairs + 4 G2-pairs) + LayerNorm -> sP ----------
  // raw-x rows additionally stored as bf16 into sX for the q GEMM
  for (int i = 0; i < 4; ++i) {
    const int r = w * 4 + i;                   // 32 rows over 8 waves
    unsigned short* dst = sP + r * CDIM + lane * 16;
    if (r >= 28) {                             // zero pad rows
      const uint4 z = make_uint4(0, 0, 0, 0);
      *(uint4*)(dst) = z; *(uint4*)(dst + 8) = z;
      continue;
    }
    float v[16];
    if (r < 16) {
      ld16(xb + r * CDIM + lane * 16, v);
      union { u16x8 h[2]; unsigned short s16[16]; } o2;
      #pragma unroll
      for (int t = 0; t < 16; ++t) o2.s16[t] = f2bf(v[t]);
      unsigned short* xd = sX + r * CDIM + lane * 16;
      *(u16x8*)(xd) = o2.h[0];
      *(u16x8*)(xd + 8) = o2.h[1];
    } else if (r < 24) {
      const int j = r - 16;
      float va[16], vb[16];
      ld16(xb + G1[2 * j] * CDIM + lane * 16, va);
      ld16(xb + G1[2 * j + 1] * CDIM + lane * 16, vb);
      #pragma unroll
      for (int t = 0; t < 16; ++t) v[t] = 0.5f * (va[t] + vb[t]);
    } else {
      const int j = r - 24;
      float va[16], vb[16];
      ld16(xb + G2[2 * j] * CDIM + lane * 16, va);
      ld16(xb + G2[2 * j + 1] * CDIM + lane * 16, vb);
      #pragma unroll
      for (int t = 0; t < 16; ++t) v[t] = 0.5f * (va[t] + vb[t]);
    }
    float s = 0.f, ss = 0.f;
    #pragma unroll
    for (int t = 0; t < 16; ++t) { s += v[t]; ss += v[t] * v[t]; }
    #pragma unroll
    for (int off = 1; off < 32; off <<= 1) { s += __shfl_xor(s, off); ss += __shfl_xor(ss, off); }
    const float mu  = s * (1.0f / 512.0f);
    const float var = fmaxf(ss * (1.0f / 512.0f) - mu * mu, 0.0f);
    const float rs  = rsqrtf(var + 1e-5f);
    float g[16], be[16];
    ld16(gamma + lane * 16, g);
    ld16(beta + lane * 16, be);
    union { u16x8 h[2]; unsigned short s16[16]; } o;
    #pragma unroll
    for (int t = 0; t < 16; ++t) o.s16[t] = f2bf((v[t] - mu) * rs * g[t] + be[t]);
    *(u16x8*)(dst) = o.h[0];
    *(u16x8*)(dst + 8) = o.h[1];
  }
  __syncthreads();

  // ---------- q = (x @ Wq)[:, h*64 : h*64+64] * 0.125, bf16 -> sQO wave region ----------
  {
    __builtin_prefetch(wqP + ((4 * w * 16) * 32 + lane) * 16, 0, 1);
    v8f aq[4] = {vzero, vzero, vzero, vzero};
    for (int kb = 0; kb < 16; ++kb) {
      const V16 A = ldsA(sX + nl * CDIM + kb * 32 + half * 8);
      #pragma unroll
      for (int j = 0; j < 4; ++j) {
        const V16 Bj = ldgB(wqP + (((4 * w + j) * 16 + kb) * 32 + lane) * 16);
        aq[j] = wmma_bf16(A, Bj, aq[j]);
      }
    }
    unsigned short* qh = sQO + w * (NTOK * HD);
    #pragma unroll
    for (int j = 0; j < 4; ++j)
      #pragma unroll
      for (int r = 0; r < 8; ++r)
        qh[(8 * half + r) * HD + j * 16 + nl] = f2bf(aq[j][r] * 0.125f);
  }

  // ---------- k_h = pools @ Wkv[:, h*64:...] (kept as bf16 regs until pbuf is dead) ----------
  unsigned short kst[4][16];
  {
    v8f ka[2][4] = {{vzero, vzero, vzero, vzero}, {vzero, vzero, vzero, vzero}};
    for (int kb = 0; kb < 16; ++kb) {
      const V16 A0 = ldsA(sP + nl * CDIM + kb * 32 + half * 8);
      const V16 A1 = ldsA(sP + (16 + nl) * CDIM + kb * 32 + half * 8);
      #pragma unroll
      for (int j = 0; j < 4; ++j) {
        const V16 Bk = ldgB(wkvP + (((4 * w + j) * 16 + kb) * 32 + lane) * 16);
        ka[0][j] = wmma_bf16(A0, Bk, ka[0][j]);
        ka[1][j] = wmma_bf16(A1, Bk, ka[1][j]);
      }
    }
    #pragma unroll
    for (int j = 0; j < 4; ++j)
      #pragma unroll
      for (int e = 0; e < 16; ++e)
        kst[j][e] = f2bf(e < 8 ? ka[0][j][e] : ka[1][j][e - 8]);
  }

  // ---------- v_h: D-layout converts in-lane to PV B-operand (never touches LDS) ----------
  V16 vB[4];
  {
    v8f va[2][4] = {{vzero, vzero, vzero, vzero}, {vzero, vzero, vzero, vzero}};
    for (int kb = 0; kb < 16; ++kb) {
      const V16 A0 = ldsA(sP + nl * CDIM + kb * 32 + half * 8);
      const V16 A1 = ldsA(sP + (16 + nl) * CDIM + kb * 32 + half * 8);
      #pragma unroll
      for (int j = 0; j < 4; ++j) {
        const V16 Bv = ldgB(wkvP + (((32 + 4 * w + j) * 16 + kb) * 32 + lane) * 16);
        va[0][j] = wmma_bf16(A0, Bv, va[0][j]);
        va[1][j] = wmma_bf16(A1, Bv, va[1][j]);
      }
    }
    #pragma unroll
    for (int j = 0; j < 4; ++j)
      #pragma unroll
      for (int e = 0; e < 16; ++e)
        vB[j].s[e] = f2bf(e < 8 ? va[0][j][e] : va[1][j][e - 8]);
  }
  __syncthreads();   // all pool reads done; sP becomes per-wave scratch

  // ---------- k_h -> row-major [token][d] in per-wave scratch (the one transpose) ----------
  unsigned short* khrm = sP + w * (MPAD * HD);   // 4KB per wave
  #pragma unroll
  for (int j = 0; j < 4; ++j)
    #pragma unroll
    for (int e = 0; e < 16; ++e) {
      const int tok = (e < 8) ? (half * 8 + e) : (16 + half * 8 + (e - 8));
      khrm[tok * HD + j * 16 + nl] = kst[j][e];
    }

  // ---------- S = q_h @ k_h^T (16 x 32, K=64), two N-tiles ----------
  v8f s0 = vzero, s1 = vzero;
  {
    const unsigned short* qh = sQO + w * (NTOK * HD);
    #pragma unroll
    for (int kb2 = 0; kb2 < 2; ++kb2) {
      const V16 A  = ldsA(qh + nl * HD + kb2 * 32 + half * 8);
      const V16 B0 = ldsA(khrm + nl * HD + kb2 * 32 + half * 8);
      const V16 B1 = ldsA(khrm + (16 + nl) * HD + kb2 * 32 + half * 8);
      s0 = wmma_bf16(A, B0, s0);
      s1 = wmma_bf16(A, B1, s1);
    }
  }

  // ---------- softmax over 32 (cols 28..31 masked), P row-major into scratch ----------
  #pragma unroll
  for (int r = 0; r < 8; ++r) {
    const float a0 = s0[r];
    const float a1 = (nl >= 12) ? -1e30f : s1[r];
    float mx = fmaxf(a0, a1);
    #pragma unroll
    for (int off = 1; off < 16; off <<= 1) mx = fmaxf(mx, __shfl_xor(mx, off));
    const float e0 = __expf(a0 - mx);
    const float e1 = (nl >= 12) ? 0.0f : __expf(a1 - mx);
    float sm = e0 + e1;
    #pragma unroll
    for (int off = 1; off < 16; off <<= 1) sm += __shfl_xor(sm, off);
    const float inv = 1.0f / sm;
    const int qt = half * 8 + r;
    khrm[qt * 32 + nl]      = f2bf(e0 * inv);
    khrm[qt * 32 + 16 + nl] = f2bf(e1 * inv);
  }

  // ---------- O_h = P @ v_h (K=32, one WMMA per d-tile; v B-operand from registers) ----------
  v8f oacc[4];
  {
    const V16 Ap = ldsA(khrm + nl * 32 + half * 8);
    #pragma unroll
    for (int ntv = 0; ntv < 4; ++ntv) oacc[ntv] = wmma_bf16(Ap, vB[ntv], vzero);
  }
  __syncthreads();   // all q reads done; sQO becomes O

  #pragma unroll
  for (int ntv = 0; ntv < 4; ++ntv)
    #pragma unroll
    for (int r = 0; r < 8; ++r)
      sQO[(8 * half + r) * CDIM + w * HD + ntv * 16 + nl] = f2bf(oacc[ntv][r]);
  __syncthreads();

  // ---------- final = O @ Wproj + bproj, wave w owns output cols [64w, 64w+64) ----------
  {
    v8f ap[4] = {vzero, vzero, vzero, vzero};
    for (int kb = 0; kb < 16; ++kb) {
      const V16 A = ldsA(sQO + nl * CDIM + kb * 32 + half * 8);
      #pragma unroll
      for (int j = 0; j < 4; ++j) {
        const V16 Bp = ldgB(wprP + (((4 * w + j) * 16 + kb) * 32 + lane) * 16);
        ap[j] = wmma_bf16(A, Bp, ap[j]);
      }
    }
    #pragma unroll
    for (int j = 0; j < 4; ++j) {
      const int col = (4 * w + j) * 16 + nl;
      const float bias = bproj[col];
      #pragma unroll
      for (int r = 0; r < 8; ++r)
        out[((size_t)b * NTOK + 8 * half + r) * CDIM + col] = ap[j][r] + bias;
    }
  }
}

extern "C" void kernel_launch(void* const* d_in, const int* in_sizes, int n_in,
                              void* d_out, int out_size, void* d_ws, size_t ws_size,
                              hipStream_t stream) {
  const float* x     = (const float*)d_in[0];
  const float* Wq    = (const float*)d_in[1];
  const float* Wkv   = (const float*)d_in[2];
  const float* Wproj = (const float*)d_in[3];
  const float* bproj = (const float*)d_in[4];
  const float* gamma = (const float*)d_in[5];
  const float* beta  = (const float*)d_in[6];
  unsigned short* wp = (unsigned short*)d_ws;   // 2 MB packed bf16 weights

  pack_weights<<<1048576 / 256, 256, 0, stream>>>(Wq, Wkv, Wproj, wp);
  attn_fused<<<4096, 256, 0, stream>>>(x, wp, bproj, gamma, beta, (float*)d_out);
}
